// MultiHeadAttention_72198400246006
// MI455X (gfx1250) — matrix operand order repaired and checked by reading, not compiled, not run on hardware
//
#include <hip/hip_runtime.h>

typedef _Float16 f16;
typedef __attribute__((ext_vector_type(16))) _Float16 v16h;
typedef __attribute__((ext_vector_type(8)))  _Float16 v8h;
typedef __attribute__((ext_vector_type(8)))  float    v8f;
typedef __attribute__((ext_vector_type(4)))  float    v4f;
typedef unsigned int u32x4 __attribute__((ext_vector_type(4)));
typedef int          i32x4 __attribute__((ext_vector_type(4)));
typedef int          i32x8 __attribute__((ext_vector_type(8)));

#define DEVI static __device__ __forceinline__

#ifndef __has_builtin
#define __has_builtin(x) 0
#endif

#if __has_builtin(__builtin_amdgcn_global_load_async_to_lds_b128) && \
    __has_builtin(__builtin_amdgcn_s_wait_asynccnt)
#define HAVE_ASYNC 1
#else
#define HAVE_ASYNC 0
#endif

#if __has_builtin(__builtin_amdgcn_tensor_load_to_lds) && \
    __has_builtin(__builtin_amdgcn_s_wait_tensorcnt)
#define HAVE_TDM 1
#else
#define HAVE_TDM 0
#endif

// ---------------- model dims ----------------
constexpr int   N_ = 8, S_ = 2048, E_ = 256, H_ = 4, P_ = 16;
constexpr int   SP  = S_ + P_;      // 2064 keys incl persistent
constexpr int   SPP = 2080;         // padded to multiple of 32
constexpr int   KB  = SPP / 32;     // 65 key blocks
constexpr float NEGVAL = -625.0f;   // -1e4 / sqrt(E)

// ---------------- workspace layout (in halves) ----------------
constexpr size_t QOFF = 0;
constexpr size_t QSZ  = (size_t)N_ * H_ * S_ * E_;          // Qtilde (N,H,S,256)
constexpr size_t KOFF = QOFF + QSZ;
constexpr size_t KSZ  = (size_t)N_ * SPP * E_;              // Ktilde (N,SPP,256)
constexpr size_t VOFF = KOFF + KSZ;                          // Vtilde^T (N,256,SPP)
constexpr size_t AOFF = VOFF + KSZ;                          // attn out (N*S,256)
constexpr size_t ASZ  = (size_t)N_ * S_ * E_;
constexpr size_t WOFF = AOFF + ASZ;                          // Wo f16 (256,256)

// ---------------- async / TDM staging helpers ----------------
DEVI void asyncCopyB128(const f16* g, f16* l) {
#if HAVE_ASYNC
  typedef __attribute__((ext_vector_type(4))) int i4;
  __builtin_amdgcn_global_load_async_to_lds_b128(
      (__attribute__((address_space(1))) i4*)(uintptr_t)g,
      (__attribute__((address_space(3))) i4*)(uintptr_t)(uint32_t)(uintptr_t)l,
      0, 0);
#else
  *(v8h*)l = *(const v8h*)g;
#endif
}

#if HAVE_TDM
// 2D tile load via Tensor Data Mover: tile (tileRows x tileK) of 2-byte
// elements from a row-major tensor with row stride strideElem, into LDS.
DEVI void tdmLoad2D(const f16* gtile, const f16* ltile, unsigned tileK,
                    unsigned tileRows, unsigned dim0, unsigned dim1,
                    unsigned strideElem) {
  unsigned long long ga = (unsigned long long)(uintptr_t)gtile;
  unsigned la = (unsigned)(uintptr_t)ltile;       // LDS byte offset (low 32b)
  u32x4 g0;
  g0.x = 1u;                                       // count=1, user descriptor
  g0.y = la;                                       // lds_addr
  g0.z = (unsigned)ga;                             // global_addr[31:0]
  g0.w = (unsigned)((ga >> 32) & 0x01FFFFFFu) | 0x80000000u;  // type=2
  i32x8 g1;
  g1[0] = 0x00010000;                              // data_size = 2 bytes
  g1[1] = (int)((dim0 & 0xFFFFu) << 16);           // tensor_dim0 lo16
  g1[2] = (int)(((dim0 >> 16) & 0xFFFFu) | ((dim1 & 0xFFFFu) << 16));
  g1[3] = (int)(((dim1 >> 16) & 0xFFFFu) | (tileK << 16));   // tile_dim0
  g1[4] = (int)tileRows;                           // tile_dim1 (tile_dim2 = 0)
  g1[5] = (int)strideElem;                         // tensor_dim0_stride lo32
  g1[6] = 0;
  g1[7] = 0;
  i32x4 z4 = {0, 0, 0, 0};
#if __clang_major__ >= 23
  i32x8 z8 = {0, 0, 0, 0, 0, 0, 0, 0};
  __builtin_amdgcn_tensor_load_to_lds(g0, g1, z4, z4, z8, 0);
#else
  __builtin_amdgcn_tensor_load_to_lds(g0, g1, z4, z4, 0);
#endif
}
#endif

// ---------------- WMMA fragment loaders (documented gfx1250 layouts) -------
DEVI v16h loadA(const f16* base, int row0, int col0, int ld, int lane) {
  const int m  = lane & 15;
  const int kb = (lane & 16) ? 8 : 0;
  const f16* p = base + (size_t)(row0 + m) * ld + col0 + kb;
  v8h lo = *(const v8h*)p;
  v8h hi = *(const v8h*)(p + 16);
  return __builtin_shufflevector(lo, hi, 0,1,2,3,4,5,6,7,8,9,10,11,12,13,14,15);
}
DEVI v16h loadB(const f16* base, int row0, int col0, int ld, int lane) {
  const int nn = lane & 15;
  const int cb = (lane & 16) ? 8 : 0;
  const f16* p = base + (size_t)(row0 + nn) * ld + col0 + cb;
  v8h lo = *(const v8h*)p;
  v8h hi = *(const v8h*)(p + 16);
  return __builtin_shufflevector(lo, hi, 0,1,2,3,4,5,6,7,8,9,10,11,12,13,14,15);
}
DEVI v8f wmma16(v16h a, v16h b, v8f c) {
  return __builtin_amdgcn_wmma_f32_16x16x32_f16(false, a, false, b, (short)0, c,
                                                false, false);
}
DEVI v8f vzero() { v8f z = {0.f,0.f,0.f,0.f,0.f,0.f,0.f,0.f}; return z; }

DEVI float redmax16(float x) {
  x = fmaxf(x, __shfl_xor(x, 1, 32));
  x = fmaxf(x, __shfl_xor(x, 2, 32));
  x = fmaxf(x, __shfl_xor(x, 4, 32));
  x = fmaxf(x, __shfl_xor(x, 8, 32));
  return x;
}
DEVI float redsum16(float x) {
  x += __shfl_xor(x, 1, 32);
  x += __shfl_xor(x, 2, 32);
  x += __shfl_xor(x, 4, 32);
  x += __shfl_xor(x, 8, 32);
  return x;
}

// ---------------- kernel: persistent K/V rows + zero padding ---------------
__global__ void k_fill(const float* __restrict__ pk, const float* __restrict__ pv,
                       f16* __restrict__ ws) {
  int idx = blockIdx.x * 256 + threadIdx.x;          // N*256*32 = 65536
  int n  = idx >> 13;
  int rem = idx & 8191;
  int c  = rem >> 5;
  int kk = rem & 31;
  int k  = S_ + kk;
  int d  = c & 63;
  float vk = (kk < P_) ? pk[kk * 64 + d] : 0.f;
  float vv = (kk < P_) ? pv[kk * 64 + d] : 0.f;
  ws[KOFF + ((size_t)n * SPP + k) * E_ + c] = (f16)vk;
  ws[VOFF + ((size_t)n * E_ + c) * SPP + k] = (f16)vv;
}

// ---------------- kernel: Wo -> f16 ----------------
__global__ void k_wconv(const float* __restrict__ Wo, f16* __restrict__ ws) {
  int idx = blockIdx.x * 256 + threadIdx.x;          // 65536
  ws[WOFF + idx] = (f16)Wo[idx];
}

// ---------------- kernel: per-head projections, pre-talking-heads fold -----
__global__ __launch_bounds__(128, 1)
void k_proj(const float* __restrict__ vals, const float* __restrict__ keys,
            const float* __restrict__ qrys, const float* __restrict__ Wv,
            const float* __restrict__ Wk,   const float* __restrict__ Wq,
            const float* __restrict__ pre_th, f16* __restrict__ ws) {
  __shared__ __align__(16) f16 Xl[16][256];
  __shared__ __align__(16) f16 Wl[64][64];

  const int tile = blockIdx.x;
  const int n    = tile >> 7;
  const int qb   = (tile & 127) << 4;
  const int msel = blockIdx.y;                 // 0=v 1=k 2=q
  const int tid  = threadIdx.x;
  const int w    = tid >> 5;                   // head j
  const int lane = tid & 31;

  const float* X = (msel == 0) ? vals : (msel == 1) ? keys : qrys;
  const float* W = (msel == 0) ? Wv   : (msel == 1) ? Wk   : Wq;

  { // stage x tile (16x256) f32->f16
    const float* src = X + (size_t)(n * S_ + qb) * E_;
    int row = tid >> 3, seg = tid & 7;
    const float* s = src + row * E_ + seg * 32;
    f16* d = &Xl[row][seg * 32];
    #pragma unroll
    for (int u = 0; u < 32; u += 4) {
      v4f f = *(const v4f*)(s + u);
      d[u + 0] = (f16)f.x; d[u + 1] = (f16)f.y;
      d[u + 2] = (f16)f.z; d[u + 3] = (f16)f.w;
    }
  }
  { // stage W (64x64) f32->f16
    int base = tid * 32;
    #pragma unroll
    for (int u = 0; u < 32; u += 4) {
      int idx = base + u;
      v4f f = *(const v4f*)(W + idx);
      f16* d = &Wl[0][0] + idx;
      d[0] = (f16)f.x; d[1] = (f16)f.y; d[2] = (f16)f.z; d[3] = (f16)f.w;
    }
  }
  __syncthreads();

  v8f acc[4];
  #pragma unroll
  for (int t = 0; t < 4; ++t) acc[t] = vzero();

  #pragma unroll
  for (int ks = 0; ks < 2; ++ks) {
    v16h a = loadA(&Xl[0][0], 0, w * 64 + ks * 32, 256, lane);
    #pragma unroll
    for (int nt = 0; nt < 4; ++nt) {
      v16h b = loadB(&Wl[0][0], nt * 16, ks * 32, 64, lane);
      acc[nt] = wmma16(a, b, acc[nt]);
    }
  }

  f16* Qg = ws + QOFF; f16* Kg = ws + KOFF; f16* Vt = ws + VOFF;
  const int rsh = (lane >> 4) << 3;
  const int cn  = lane & 15;

  if (msel == 2) {                 // Qtilde: fold pre_th[i,j]/sqrt(E)
    float pw[4];
    #pragma unroll
    for (int i = 0; i < 4; ++i) pw[i] = pre_th[i * H_ + w] * 0.0625f;
    #pragma unroll
    for (int nt = 0; nt < 4; ++nt) {
      #pragma unroll
      for (int r = 0; r < 8; ++r) {
        int q  = qb + r + rsh;
        int dd = w * 64 + nt * 16 + cn;
        float v = acc[nt][r];
        #pragma unroll
        for (int i = 0; i < 4; ++i)
          Qg[((size_t)(n * H_ + i) * S_ + q) * E_ + dd] = (f16)(v * pw[i]);
      }
    }
  } else if (msel == 1) {          // Ktilde
    #pragma unroll
    for (int nt = 0; nt < 4; ++nt)
      #pragma unroll
      for (int r = 0; r < 8; ++r)
        Kg[((size_t)n * SPP + qb + r + rsh) * E_ + w * 64 + nt * 16 + cn] =
            (f16)acc[nt][r];
  } else {                         // Vtilde^T
    #pragma unroll
    for (int nt = 0; nt < 4; ++nt)
      #pragma unroll
      for (int r = 0; r < 8; ++r)
        Vt[((size_t)n * E_ + w * 64 + nt * 16 + cn) * SPP + (qb + r + rsh)] =
            (f16)acc[nt][r];
  }
}

// ---------------- kernel: fused flash attention with talking heads --------
__global__ __launch_bounds__(128, 1)
void k_attn(const int* __restrict__ mask, const float* __restrict__ pre_th,
            const float* __restrict__ post_th, f16* __restrict__ ws) {
  __shared__ __align__(16) f16 Ql[4][16][256];    // 32 KB
  __shared__ __align__(16) f16 Kl[2][32][256];    // 32 KB double buffer
  __shared__ __align__(16) f16 Vl[2][256][32];    // 32 KB double buffer
  __shared__ __align__(16) f16 Pl[4][16][32];     //  4 KB

  const int n   = blockIdx.x >> 7;
  const int qb  = (blockIdx.x & 127) << 4;
  const int tid = threadIdx.x, w = tid >> 5, lane = tid & 31;

  const f16* Qg = ws + QOFF;
  const f16* Kg = ws + KOFF;
  const f16* Vt = ws + VOFF;
  f16*       Ag = ws + AOFF;

  // pre-mixed ALiBi slopes (includes 1/sqrt(E))
  float slope[4];
  #pragma unroll
  for (int i = 0; i < 4; ++i) {
    float s = 0.f;
    #pragma unroll
    for (int j = 0; j < 4; ++j) s += pre_th[i * 4 + j] * exp2f(-(float)(j + 1));
    slope[i] = s * 0.0625f;
  }

  // stage Qtilde: 4 heads x 16 x 256 halves (plain path, once per block)
  #pragma unroll
  for (int it = 0; it < 16; ++it) {
    int chunk = tid + 128 * it;
    int i = chunk >> 9, rem = chunk & 511, r = rem >> 5, cs = rem & 31;
    *(v8h*)&Ql[i][r][cs * 8] =
        *(const v8h*)(Qg + ((size_t)(n * H_ + i) * S_ + qb + r) * E_ + cs * 8);
  }

  // staging issue helpers (double-buffered)
  auto issueK = [&](int buf, int kb) {
    #pragma unroll
    for (int it = 0; it < 8; ++it) {
      int chunk = tid + 128 * it;
      int r = chunk >> 5, cs = chunk & 31;
      asyncCopyB128(Kg + ((size_t)n * SPP + kb + r) * E_ + cs * 8,
                    &Kl[buf][r][cs * 8]);
    }
  };
  auto issueV = [&](int buf, int kb) {
#if HAVE_TDM
    if (w == 0)
      tdmLoad2D(Vt + (size_t)n * E_ * SPP + kb, &Vl[buf][0][0],
                32u, 256u, (unsigned)SPP, (unsigned)E_, (unsigned)SPP);
#else
    #pragma unroll
    for (int it = 0; it < 8; ++it) {
      int chunk = tid + 128 * it;
      int r = chunk >> 2, cs = chunk & 3;
      asyncCopyB128(Vt + ((size_t)n * E_ + r) * SPP + kb + cs * 8,
                    &Vl[buf][r][cs * 8]);
    }
#endif
  };

  v8f   acc[4][4];
  float mr[4][8], lr[4][8];
  #pragma unroll
  for (int i = 0; i < 4; ++i) {
    #pragma unroll
    for (int dt = 0; dt < 4; ++dt) acc[i][dt] = vzero();
    #pragma unroll
    for (int r = 0; r < 8; ++r) { mr[i][r] = -3.0e38f; lr[i][r] = 0.f; }
  }

  const int rsh = (lane >> 4) << 3;
  const int cn  = lane & 15;
  float qf[8];
  #pragma unroll
  for (int r = 0; r < 8; ++r) qf[r] = (float)(qb + r + rsh);

  issueK(0, 0);
  issueV(0, 0);

  #pragma unroll 1
  for (int kbi = 0; kbi < KB; ++kbi) {
    const int kb  = kbi * 32;
    const int cur = kbi & 1;
#if HAVE_TDM
    if (w == 0) __builtin_amdgcn_s_wait_tensorcnt(0);
#endif
#if HAVE_ASYNC
    __builtin_amdgcn_s_wait_asynccnt(0);
#endif
    __syncthreads();
    if (kbi + 1 < KB) { issueK(1 - cur, kb + 32); issueV(1 - cur, kb + 32); }

    const f16* Klc = &Kl[cur][0][0];
    const f16* Vlc = &Vl[cur][0][0];

    // V B-fragments for this wave's output head
    v16h bv[4];
    #pragma unroll
    for (int dt = 0; dt < 4; ++dt)
      bv[dt] = loadB(Vlc, w * 64 + dt * 16, 0, 32, lane);

    // scores for all 4 softmax heads, sharing K fragments
    v8f e[4][2];
    #pragma unroll
    for (int i = 0; i < 4; ++i) { e[i][0] = vzero(); e[i][1] = vzero(); }
    #pragma unroll
    for (int ks = 0; ks < 8; ++ks) {
      v16h b0 = loadB(Klc, 0,  ks * 32, 256, lane);
      v16h b1 = loadB(Klc, 16, ks * 32, 256, lane);
      #pragma unroll
      for (int i = 0; i < 4; ++i) {
        v16h a = loadA(&Ql[i][0][0], 0, ks * 32, 256, lane);
        e[i][0] = wmma16(a, b0, e[i][0]);
        e[i][1] = wmma16(a, b1, e[i][1]);
      }
    }

    // head-independent bias/mask terms
    float ndist[2][8];
    bool  bad[2];
    #pragma unroll
    for (int t = 0; t < 2; ++t) {
      int kg = kb + t * 16 + cn;
      int mv = (kg < S_) ? mask[n * S_ + kg] : 1;
      bad[t] = (kg >= SP) || ((kg < S_) && (mv == 0));
      float kgf = (float)kg;
      bool  inS = kg < S_;
      #pragma unroll
      for (int r = 0; r < 8; ++r)
        ndist[t][r] = inS ? -fabsf(qf[r] - kgf) : 0.f;
    }

    #pragma unroll
    for (int i = 0; i < 4; ++i) {
      v8f e0 = e[i][0], e1 = e[i][1];
      #pragma unroll
      for (int r = 0; r < 8; ++r) {
        float v0 = e0[r] + ndist[0][r] * slope[i];
        float v1 = e1[r] + ndist[1][r] * slope[i];
        e0[r] = bad[0] ? NEGVAL : v0;
        e1[r] = bad[1] ? NEGVAL : v1;
      }
      float mnew[8], rf[8];
      #pragma unroll
      for (int r = 0; r < 8; ++r) {
        float bm = redmax16(fmaxf(e0[r], e1[r]));
        mnew[r]  = fmaxf(mr[i][r], bm);
        rf[r]    = __expf(mr[i][r] - mnew[r]);
        mr[i][r] = mnew[r];
      }
      #pragma unroll
      for (int r = 0; r < 8; ++r) {
        e0[r] = __expf(e0[r] - mnew[r]);
        e1[r] = __expf(e1[r] - mnew[r]);
        float rs = redsum16(e0[r] + e1[r]);
        lr[i][r] = lr[i][r] * rf[r] + rs;
      }
      #pragma unroll
      for (int dt = 0; dt < 4; ++dt)
        #pragma unroll
        for (int r = 0; r < 8; ++r)
          acc[i][dt][r] *= rf[r];
      // p -> LDS f16 (wave-private), reload as A fragment
      #pragma unroll
      for (int r = 0; r < 8; ++r) {
        Pl[w][r + rsh][cn]      = (f16)e0[r];
        Pl[w][r + rsh][16 + cn] = (f16)e1[r];
      }
      v16h ap = loadA(&Pl[w][0][0], 0, 0, 32, lane);
      #pragma unroll
      for (int dt = 0; dt < 4; ++dt)
        acc[i][dt] = wmma16(ap, bv[dt], acc[i][dt]);
    }
  }

  // finalize: out_o = sum_i post[o,i] * acc_i / l_i
  float il[4][8];
  #pragma unroll
  for (int i = 0; i < 4; ++i) {
    float pw = post_th[w * 4 + i];
    #pragma unroll
    for (int r = 0; r < 8; ++r) il[i][r] = pw / lr[i][r];
  }
  #pragma unroll
  for (int dt = 0; dt < 4; ++dt) {
    #pragma unroll
    for (int r = 0; r < 8; ++r) {
      float o = 0.f;
      #pragma unroll
      for (int i = 0; i < 4; ++i) o += il[i][r] * acc[i][dt][r];
      Ag[((size_t)(n * S_ + qb + r + rsh)) * E_ + w * 64 + dt * 16 + cn] = (f16)o;
    }
  }
}

// ---------------- kernel: output projection att @ Wo^T + bo ----------------
__global__ __launch_bounds__(128, 1)
void k_out(const float* __restrict__ bo, const f16* __restrict__ ws,
           float* __restrict__ out) {
  __shared__ __align__(16) f16 Al[16][256];
  const f16* Ag   = ws + AOFF;
  const f16* Wo16 = ws + WOFF;
  const int row0 = blockIdx.x << 4;
  const int tid  = threadIdx.x, w = tid >> 5, lane = tid & 31;

  #pragma unroll
  for (int it = 0; it < 4; ++it) {
    int chunk = tid + 128 * it;
    int r = chunk >> 5, cs = chunk & 31;
    asyncCopyB128(Ag + (size_t)(row0 + r) * E_ + cs * 8, &Al[r][cs * 8]);
  }
#if HAVE_ASYNC
  __builtin_amdgcn_s_wait_asynccnt(0);
#endif
  __syncthreads();

  v8f acc[4];
  #pragma unroll
  for (int t = 0; t < 4; ++t) acc[t] = vzero();

  #pragma unroll
  for (int ks = 0; ks < 8; ++ks) {
    v16h a = loadA(&Al[0][0], 0, ks * 32, 256, lane);
    #pragma unroll
    for (int nt = 0; nt < 4; ++nt) {
      v16h b = loadB(Wo16, w * 64 + nt * 16, ks * 32, 256, lane);
      acc[nt] = wmma16(a, b, acc[nt]);
    }
  }
  const int rsh = (lane >> 4) << 3, cn = lane & 15;
  #pragma unroll
  for (int nt = 0; nt < 4; ++nt) {
    int   e  = w * 64 + nt * 16 + cn;
    float bb = bo[e];
    #pragma unroll
    for (int r = 0; r < 8; ++r)
      out[(size_t)(row0 + r + rsh) * E_ + e] = acc[nt][r] + bb;
  }
}

// ---------------- host launcher ----------------
extern "C" void kernel_launch(void* const* d_in, const int* in_sizes, int n_in,
                              void* d_out, int out_size, void* d_ws, size_t ws_size,
                              hipStream_t stream) {
  (void)in_sizes; (void)n_in; (void)out_size; (void)ws_size;
  const float* vals = (const float*)d_in[0];
  const float* keys = (const float*)d_in[1];
  const float* qrys = (const float*)d_in[2];
  const int*   mask = (const int*)d_in[3];
  const float* Wv   = (const float*)d_in[4];
  const float* Wk   = (const float*)d_in[5];
  const float* Wq   = (const float*)d_in[6];
  const float* Wo   = (const float*)d_in[7];
  const float* bo   = (const float*)d_in[8];
  const float* pre  = (const float*)d_in[9];
  const float* post = (const float*)d_in[10];
  const float* pk   = (const float*)d_in[11];
  const float* pv   = (const float*)d_in[12];
  f16*   ws  = (f16*)d_ws;
  float* out = (float*)d_out;

  k_fill <<<256, 256, 0, stream>>>(pk, pv, ws);
  k_wconv<<<256, 256, 0, stream>>>(Wo, ws);
  k_proj <<<dim3(N_ * S_ / 16, 3), 128, 0, stream>>>(vals, keys, qrys,
                                                     Wv, Wk, Wq, pre, ws);
  k_attn <<<N_ * S_ / 16, 128, 0, stream>>>(mask, pre, post, ws);
  k_out  <<<N_ * S_ / 16, 128, 0, stream>>>(bo, ws, out);
}
